// BlockSparseAttentionHead_9844065042850
// MI455X (gfx1250) — matrix operand order repaired and checked by reading, not compiled, not run on hardware
//
#include <hip/hip_runtime.h>

// Block-diagonal (local) attention head, fully fused per (batch, block) tile.
// B=8, S=4096, D=1024, Dh=128, BLOCK=128 -> 256 workgroups of 8 wave32s.
// All matrix math via v_wmma_f32_16x16x32_bf16.

#define BATCH 8
#define SEQ   4096
#define DIM   1024
#define DHD   128
#define BLK   128
#define NBLK  (SEQ / BLK)   // 32
#define KCH   32            // WMMA K per step (bf16)
#define PITCH 136           // LDS row pitch (elements) for 128-wide bf16 tiles

typedef __attribute__((ext_vector_type(16))) __bf16 v16bf;
typedef __attribute__((ext_vector_type(4)))  __bf16 bf16x4;
typedef __attribute__((ext_vector_type(8)))  float  v8f;
typedef __attribute__((ext_vector_type(4)))  float  fx4;   // clang ext-vector: OK for nontemporal builtins

union Frag  { uint4  u4[2]; v16bf v; };
union Pack4 { bf16x4 h; uint2 u; };
union Pack1 { __bf16 h; unsigned short u; };

// hardware f32 -> bf16 (RNE) convert; single VALU op on gfx1250
__device__ __forceinline__ unsigned short bf_bits(float f) {
  Pack1 p; p.h = (__bf16)f; return p.u;
}

__device__ __forceinline__ v16bf load_frag(const unsigned short* p0,
                                           const unsigned short* p1) {
  Frag f;
  f.u4[0] = *(const uint4*)p0;          // 8 bf16 (16B aligned by construction)
  f.u4[1] = *(const uint4*)p1;          // 8 bf16
  return f.v;
}

#define WMMA_BF16(A, B, C) \
  __builtin_amdgcn_wmma_f32_16x16x32_bf16(false, (A), false, (B), (short)0, (C), false, false)

__global__ __launch_bounds__(256)
void blocksparse_attn_fused(const float* __restrict__ x,
                            const float* __restrict__ Wq,
                            const float* __restrict__ Wk,
                            const float* __restrict__ Wv,
                            float* __restrict__ out) {
  extern __shared__ char smem[];
  unsigned short* XS = (unsigned short*)smem;             // [128][32]  x chunk (bf16)
  unsigned short* WT = XS + 128 * 32;                     // [3][128 col][32 k] W chunks, K-transposed
  unsigned short* QS = (unsigned short*)(smem + 32768);   // [128][PITCH] Q tile; later reused for P
  unsigned short* KS = QS + 128 * PITCH;                  // [128][PITCH] K tile (row-major: [token][dh])
  unsigned short* VT = KS + 128 * PITCH;                  // [128][PITCH] V tile transposed: [dh][token]

  const int tid  = threadIdx.x;
  const int wave = tid >> 5;
  const int lane = tid & 31;
  const int ln   = lane & 15;           // column-within-tile (B/C layouts), row (A layout)
  const int hf   = lane >> 4;           // lane half selector

  const int wgid = blockIdx.x;
  const int b    = wgid / NBLK;
  const int blk  = wgid % NBLK;

  const float* xb = x + ((size_t)b * SEQ + (size_t)blk * BLK) * DIM;

  v8f vzero = {0.f,0.f,0.f,0.f,0.f,0.f,0.f,0.f};
  v8f accQ[8], accK[8], accV[8];
#pragma unroll
  for (int i = 0; i < 8; ++i) { accQ[i] = vzero; accK[i] = vzero; accV[i] = vzero; }

  // ---------------- Phase 1: Q/K/V projections (128x1024 @ 1024x128, x3) ----------------
  for (int kc = 0; kc < DIM; kc += KCH) {
    // stage x chunk [128 rows][32 k] -> bf16, fx4 NT loads, packed b64 LDS stores
#pragma unroll
    for (int i = 0; i < 4; ++i) {
      int q = tid + i * 256;                     // quad index, 0..1023
      int r = q >> 3, c = (q & 7) * 4;           // row 0..127, col 0..28 step 4
      fx4 v = __builtin_nontemporal_load((const fx4*)(xb + (size_t)r * DIM + kc + c));
      Pack4 p;
      p.h = (bf16x4){ (__bf16)v.x, (__bf16)v.y, (__bf16)v.z, (__bf16)v.w };
      *(uint2*)(XS + r * 32 + c) = p.u;
    }
    // stage W chunks [32 k][128 col] -> LDS transposed [col][k]; fx4 loads, scatter b16
    const float* Ws[3] = { Wq, Wk, Wv };
#pragma unroll
    for (int m = 0; m < 3; ++m) {
      const float* W = Ws[m];
#pragma unroll
      for (int i = 0; i < 4; ++i) {
        int q = tid + i * 256;                   // quad index, 0..1023
        int r = q >> 5, c = (q & 31) * 4;        // r = k (0..31), c = out col (0..124)
        fx4 v = *(const fx4*)(W + (size_t)(kc + r) * DHD + c);
        unsigned short* wp = WT + m * 4096 + c * 32 + r;
        wp[0]   = bf_bits(v.x);
        wp[32]  = bf_bits(v.y);
        wp[64]  = bf_bits(v.z);
        wp[96]  = bf_bits(v.w);
      }
    }
    __syncthreads();

    // A fragment: rows 16*wave..+15 of x chunk.
    // A layout: lane<16 -> K 0..15 ; lane>=16 -> K 16..31
    const unsigned short* xrow = XS + (16 * wave + ln) * 32;
    v16bf afrag = load_frag(xrow + hf * 16, xrow + hf * 16 + 8);

#pragma unroll
    for (int nt = 0; nt < 8; ++nt) {
      // B layout: lane<16 -> K 0..15, lane>=16 -> K 16..31; col = ln
      const unsigned short* wp = WT + (nt * 16 + ln) * 32 + hf * 16;
      v16bf bq = load_frag(wp,        wp + 8);
      v16bf bk = load_frag(wp + 4096, wp + 4096 + 8);
      v16bf bv = load_frag(wp + 8192, wp + 8192 + 8);
      accQ[nt] = WMMA_BF16(afrag, bq, accQ[nt]);
      accK[nt] = WMMA_BF16(afrag, bk, accK[nt]);
      accV[nt] = WMMA_BF16(afrag, bv, accV[nt]);
    }
    __syncthreads();
  }

  // Spill Q/K (row-major) and V (dh-major) tiles to LDS as bf16.
  // C layout: lane<16 holds M=r, lane>=16 holds M=8+r; N = ln.
#pragma unroll
  for (int nt = 0; nt < 8; ++nt) {
    int col = nt * 16 + ln;
#pragma unroll
    for (int r = 0; r < 8; ++r) {
      int row = 16 * wave + 8 * hf + r;
      QS[row * PITCH + col] = bf_bits(accQ[nt][r]);
      KS[row * PITCH + col] = bf_bits(accK[nt][r]);
      VT[col * PITCH + row] = bf_bits(accV[nt][r]);   // transposed store
    }
  }
  __syncthreads();

  // ---------------- Phase 2a: S = Q @ K^T (128x128, K-dim = Dh = 128) ----------------
  v8f accS[8];
#pragma unroll
  for (int i = 0; i < 8; ++i) accS[i] = vzero;

#pragma unroll
  for (int kc = 0; kc < DHD; kc += KCH) {
    const unsigned short* qrow = QS + (16 * wave + ln) * PITCH + kc;
    v16bf afrag = load_frag(qrow + hf * 16, qrow + hf * 16 + 8);
#pragma unroll
    for (int nt = 0; nt < 8; ++nt) {
      // B = K^T: element (k=dh, n=token) = KS[token][dh] -> contiguous over dh
      const unsigned short* kp = KS + (nt * 16 + ln) * PITCH + kc + hf * 16;
      v16bf bfrag = load_frag(kp, kp + 8);
      accS[nt] = WMMA_BF16(afrag, bfrag, accS[nt]);
    }
  }

  // Softmax over 128 keys per row. Row M<8 lives in lanes 0-15, M>=8 in 16-31,
  // so masks 1..8 of shfl_xor reduce exactly over one row's 16 lanes.
  const float scale = 0.08838834764831845f;   // 1/sqrt(128)
  float rl[8];
#pragma unroll
  for (int r = 0; r < 8; ++r) {
    float m = -3.4e38f;
#pragma unroll
    for (int nt = 0; nt < 8; ++nt) m = fmaxf(m, accS[nt][r]);
#pragma unroll
    for (int off = 1; off < 16; off <<= 1) m = fmaxf(m, __shfl_xor(m, off, 32));
    float s = 0.f;
#pragma unroll
    for (int nt = 0; nt < 8; ++nt) {
      float p = __expf((accS[nt][r] - m) * scale);   // scale>0 commutes with max
      accS[nt][r] = p;
      s += p;
    }
#pragma unroll
    for (int off = 1; off < 16; off <<= 1) s += __shfl_xor(s, off, 32);
    rl[r] = 1.0f / s;
  }

  // Write P (bf16) over the dead Q tile.
#pragma unroll
  for (int nt = 0; nt < 8; ++nt) {
    int col = nt * 16 + ln;
#pragma unroll
    for (int r = 0; r < 8; ++r) {
      int row = 16 * wave + 8 * hf + r;
      QS[row * PITCH + col] = bf_bits(accS[nt][r]);
    }
  }
  __syncthreads();

  // ---------------- Phase 2b: O = (P @ V) * (1/rowsum) ----------------
  v8f accO[8];
#pragma unroll
  for (int i = 0; i < 8; ++i) accO[i] = vzero;

#pragma unroll
  for (int kc = 0; kc < BLK; kc += KCH) {
    const unsigned short* prow = QS + (16 * wave + ln) * PITCH + kc;
    v16bf afrag = load_frag(prow + hf * 16, prow + hf * 16 + 8);
#pragma unroll
    for (int nt = 0; nt < 8; ++nt) {
      // B element (k=token, n=dh) = VT[dh][token] -> contiguous over token
      const unsigned short* vp = VT + (nt * 16 + ln) * PITCH + kc + hf * 16;
      v16bf bfrag = load_frag(vp, vp + 8);
      accO[nt] = WMMA_BF16(afrag, bfrag, accO[nt]);
    }
  }

  float* ob = out + ((size_t)b * SEQ + (size_t)blk * BLK) * DHD;
#pragma unroll
  for (int nt = 0; nt < 8; ++nt) {
    int col = nt * 16 + ln;
#pragma unroll
    for (int r = 0; r < 8; ++r) {
      int row = 16 * wave + 8 * hf + r;
      __builtin_nontemporal_store(accO[nt][r] * rl[r], ob + (size_t)row * DHD + col);
    }
  }
}

extern "C" void kernel_launch(void* const* d_in, const int* in_sizes, int n_in,
                              void* d_out, int out_size, void* d_ws, size_t ws_size,
                              hipStream_t stream) {
  const float* x  = (const float*)d_in[0];
  const float* wq = (const float*)d_in[1];
  const float* wk = (const float*)d_in[2];
  const float* wv = (const float*)d_in[3];
  float* out = (float*)d_out;

  constexpr size_t SMEM = 32768 + (size_t)3 * 128 * PITCH * 2;  // 137,216 bytes
  dim3 grid(BATCH * NBLK);   // 256 workgroups
  dim3 block(256);           // 8 wave32s
  blocksparse_attn_fused<<<grid, block, SMEM, stream>>>(x, wq, wk, wv, out);
}
